// SuperpointGraph_64965675319802
// MI455X (gfx1250) — compile-verified, matrix operand order checked by reading, not run on hardware
//
#include <hip/hip_runtime.h>
#include <hip/hip_bf16.h>

typedef _Float16 h16;
typedef __attribute__((ext_vector_type(16))) _Float16 v16h;
typedef __attribute__((ext_vector_type(8)))  _Float16 v8h;
typedef __attribute__((ext_vector_type(8)))  float    v8f;

#define WAVE_SYNC() asm volatile("s_wait_dscnt 0" ::: "memory")

static constexpr int Bn = 2, Nn = 768, Cc = 64, Oo = 64, Ee = 18;

// Combine two contiguous 16B LDS chunks into one v16h WMMA operand.
__device__ __forceinline__ v16h ld16(const h16* p0, const h16* p1) {
  v8h lo = *(const v8h*)p0;
  v8h hi = *(const v8h*)p1;
  return __builtin_shufflevector(lo, hi, 0, 1, 2, 3, 4, 5, 6, 7,
                                 8, 9, 10, 11, 12, 13, 14, 15);
}
// A (16xK f16, row-major [16][cols]): lane m = lane&15,
// halves e<8 -> K = k0+8*hif+e ; e>=8 -> K = k0+8*hif+16+(e-8)
__device__ __forceinline__ v16h ldsA(const h16* base, int cols, int k0, int lane) {
  const int m = lane & 15, hif = lane >> 4;
  const h16* p = base + m * cols + k0 + 8 * hif;
  return ld16(p, p + 16);
}
// B (KxN f16 stored TRANSPOSED in LDS as [N][Kdim], K contiguous): lane n = n0+(lane&15)
__device__ __forceinline__ v16h ldsBt(const h16* baseT, int Kdim, int k0, int n0, int lane) {
  const int n = n0 + (lane & 15), hif = lane >> 4;
  const h16* p = baseT + n * Kdim + k0 + 8 * hif;
  return ld16(p, p + 16);
}
__device__ __forceinline__ v8f wmma_f16(v16h a, v16h b, v8f c) {
  return __builtin_amdgcn_wmma_f32_16x16x32_f16(false, a, false, b, (short)0, c, false, false);
}
__device__ __forceinline__ v8f splat8(float x) {
  v8f c;
#pragma unroll
  for (int v = 0; v < 8; ++v) c[v] = x;
  return c;
}

// ---- Stage 1: per-node projections ----
__global__ __launch_bounds__(256)
void sg_pre(const float* __restrict__ x,
            const float* __restrict__ Ws, const float* __restrict__ bs,
            const float* __restrict__ Wn, const float* __restrict__ bn,
            const float* __restrict__ Wa1, const float* __restrict__ Wg1,
            float* __restrict__ sf, float* __restrict__ nt,
            float* __restrict__ xai, float* __restrict__ xaj,
            float* __restrict__ xg) {
  __shared__ float xr[64];
  const int row = blockIdx.x;
  const int t = threadIdx.x;
  if (t < 64) xr[t] = x[(size_t)row * 64 + t];
  __syncthreads();
  if (t < 64) {
    const int o = t; float s = bs[o];
#pragma unroll
    for (int k = 0; k < 64; ++k) s += xr[k] * Ws[k * 64 + o];
    sf[(size_t)row * 64 + o] = s;
  } else if (t < 128) {
    const int o = t - 64; float s = bn[o];
#pragma unroll
    for (int k = 0; k < 64; ++k) s += xr[k] * Wn[k * 64 + o];
    nt[(size_t)row * 64 + o] = s;
  } else if (t < 160) {
    const int o = t - 128; float s = 0.f;
#pragma unroll
    for (int k = 0; k < 64; ++k) s += xr[k] * Wa1[k * 32 + o];
    xai[(size_t)row * 32 + o] = s;
  } else if (t < 192) {
    const int o = t - 160; float s = 0.f;
#pragma unroll
    for (int k = 0; k < 64; ++k) s += xr[k] * Wa1[(64 + k) * 32 + o];
    xaj[(size_t)row * 32 + o] = s;
  } else {
    const int o = t - 192; float s = 0.f;
#pragma unroll
    for (int k = 0; k < 64; ++k) s += xr[k] * Wg1[k * 64 + o];
    xg[(size_t)row * 64 + o] = s;
  }
}

// ---- Stage 2: fused per-(b,i) edge MLP + attention + gating + combine ----
__global__ __launch_bounds__(128)
void sg_main(const float* __restrict__ adj, const float* __restrict__ ef,
             const float* __restrict__ We1, const float* __restrict__ be1,
             const float* __restrict__ We2, const float* __restrict__ be2,
             const float* __restrict__ Wa1, const float* __restrict__ ba1,
             const float* __restrict__ Wa2, const float* __restrict__ ba2,
             const float* __restrict__ Wg1, const float* __restrict__ bg1,
             const float* __restrict__ Wg2, const float* __restrict__ bg2,
             const float* __restrict__ Wc1, const float* __restrict__ bc1,
             const float* __restrict__ Wc2, const float* __restrict__ bc2,
             const float* __restrict__ sf, const float* __restrict__ nt,
             const float* __restrict__ xai, const float* __restrict__ xaj,
             const float* __restrict__ xg, float* __restrict__ out) {
  // Weights stored TRANSPOSED: Wt[n*K + k]
  __shared__ alignas(16) h16 We1t[32 * 32], We2t[32 * 32], Waet[32 * 32];
  __shared__ alignas(16) h16 Wget[64 * 32], Wg2t[64 * 64];
  __shared__ float be1s[32], be2s[32], ba1s[32], wa2s[32];
  __shared__ float bg1s[64], bg2s[64], xais[32], xgs[64], sfs[64];
  __shared__ alignas(16) h16 bufE[4][16 * 32];   // E tile (pad cols zeroed once)
  __shared__ alignas(16) h16 bufH[4][16 * 32];   // e1 tile, then h tile
  __shared__ alignas(16) h16 bufPe[4][16 * 32];  // pe tile
  __shared__ alignas(16) h16 bufG[4][16 * 64];   // g tile
  __shared__ float lg[4][16];
  __shared__ float accW[4 * 64];
  __shared__ float Mw[4], Sw[4];
  __shared__ float msgv[64], t1v[64];

  const int i = blockIdx.x, b = blockIdx.y;
  const int tid = threadIdx.x, wv = tid >> 5, lane = tid & 31;
  const int nloc = lane & 15, hif = lane >> 4;  // 0 or 1
  const size_t row_i = (size_t)(b * Nn + i);

  // --- weights (transposed) / biases / per-i vectors into LDS ---
  for (int idx = tid; idx < 32 * 32; idx += 128) {
    const int n = idx >> 5, k = idx & 31;  // [n][k]
    We1t[idx] = (h16)(k < Ee ? We1[k * 32 + n] : 0.f);
    We2t[idx] = (h16)We2[k * 32 + n];
    Waet[idx] = (h16)Wa1[(2 * Cc + k) * 32 + n];
  }
  for (int idx = tid; idx < 64 * 32; idx += 128) {
    const int n = idx >> 5, k = idx & 31;  // [n][k], K=32
    Wget[idx] = (h16)Wg1[(Cc + k) * 64 + n];
  }
  for (int idx = tid; idx < 64 * 64; idx += 128) {
    const int n = idx >> 6, k = idx & 63;  // [n][k], K=64
    Wg2t[idx] = (h16)Wg2[k * 64 + n];
  }
  if (tid < 32) {
    be1s[tid] = be1[tid]; be2s[tid] = be2[tid]; ba1s[tid] = ba1[tid];
    wa2s[tid] = Wa2[tid]; xais[tid] = xai[row_i * 32 + tid];
  } else if (tid < 96) {
    const int o = tid - 32;
    bg1s[o] = bg1[o]; bg2s[o] = bg2[o];
    xgs[o] = xg[row_i * 64 + o]; sfs[o] = sf[row_i * 64 + o];
  }
  for (int idx = tid; idx < 4 * 64; idx += 128) accW[idx] = 0.f;
  // zero E-tile pad columns once (inner loop only writes k<18)
  for (int idx = lane; idx < 16 * 32; idx += 32) bufE[wv][idx] = (h16)0.f;
  __syncthreads();

  // --- per-wave online-softmax state ---
  float acc0 = 0.f, acc1 = 0.f, acc2 = 0.f, acc3 = 0.f;
  float Mrun = -1e30f, Srun = 0.f;
  const float ba2v = ba2[0];
  h16* bE = bufE[wv]; h16* bH = bufH[wv]; h16* bPe = bufPe[wv]; h16* bG = bufG[wv];

  for (int jt = wv; jt < Nn / 16; jt += 4) {
    const int j0 = jt * 16;
    // E tile: 16 rows x 18 f32 = contiguous 288 f32 block; 9 per lane.
    {
      const float* efb = ef + (row_i * Nn + j0) * Ee;
#pragma unroll
      for (int q = 0; q < 9; ++q) {
        const int flat = q * 32 + lane;       // 0..287
        const int m = flat / Ee, k = flat % Ee;
        bE[m * 32 + k] = (h16)efb[flat];
      }
    }
    WAVE_SYNC();
    // GEMM1: e1 = relu(E @ We1 + be1) -> bH (16x32); bias folded into C
    {
      const v16h a = ldsA(bE, 32, 0, lane);
#pragma unroll
      for (int t = 0; t < 2; ++t) {
        const v16h bb = ldsBt(We1t, 32, 0, 16 * t, lane);
        const int n = 16 * t + nloc;
        v8f c = wmma_f16(a, bb, splat8(be1s[n]));
#pragma unroll
        for (int v = 0; v < 8; ++v) {
          const int m = v + 8 * hif;
          const float val = c[v];
          bH[m * 32 + n] = (h16)(val > 0.f ? val : 0.f);
        }
      }
    }
    WAVE_SYNC();
    // GEMM2: pe = relu(e1 @ We2 + be2) -> bPe (16x32)
    {
      const v16h a = ldsA(bH, 32, 0, lane);
#pragma unroll
      for (int t = 0; t < 2; ++t) {
        const v16h bb = ldsBt(We2t, 32, 0, 16 * t, lane);
        const int n = 16 * t + nloc;
        v8f c = wmma_f16(a, bb, splat8(be2s[n]));
#pragma unroll
        for (int v = 0; v < 8; ++v) {
          const int m = v + 8 * hif;
          const float val = c[v];
          bPe[m * 32 + n] = (h16)(val > 0.f ? val : 0.f);
        }
      }
    }
    WAVE_SYNC();
    // GEMM3: h = relu(pe @ Wa_e + xai(i) + xaj(j) + ba1) -> bH (16x32)
    // additive terms loaded directly into the WMMA C operand
    {
      const v16h a = ldsA(bPe, 32, 0, lane);
#pragma unroll
      for (int t = 0; t < 2; ++t) {
        const v16h bb = ldsBt(Waet, 32, 0, 16 * t, lane);
        const int n = 16 * t + nloc;
        const float add_n = ba1s[n] + xais[n];
        v8f ci;
#pragma unroll
        for (int v = 0; v < 8; ++v) {
          const int m = v + 8 * hif;
          ci[v] = add_n + xaj[((size_t)(b * Nn) + (j0 + m)) * 32 + n];
        }
        v8f c = wmma_f16(a, bb, ci);
#pragma unroll
        for (int v = 0; v < 8; ++v) {
          const int m = v + 8 * hif;
          const float val = c[v];
          bH[m * 32 + n] = (h16)(val > 0.f ? val : 0.f);
        }
      }
    }
    WAVE_SYNC();
    // logits row-wise: lanes 0..15 -> lg[wv][jj]; h row read as 4x b128
    if (lane < 16) {
      const v8h* hr = (const v8h*)(bH + lane * 32);
      float s = ba2v;
#pragma unroll
      for (int cch = 0; cch < 4; ++cch) {
        const v8h hv = hr[cch];
#pragma unroll
        for (int e = 0; e < 8; ++e) s += (float)hv[e] * wa2s[cch * 8 + e];
      }
      const bool msk = adj[row_i * Nn + (j0 + lane)] > 0.f;
      lg[wv][lane] = msk ? s : -1e30f;
    }
    WAVE_SYNC();
    // online softmax update (uniform across lanes); exps computed once
    float wr[8];
    {
      float lv[16];
#pragma unroll
      for (int k = 0; k < 16; ++k) lv[k] = lg[wv][k];
      float mt = -1e30f;
#pragma unroll
      for (int k = 0; k < 16; ++k) mt = fmaxf(mt, lv[k]);
      const float newM = fmaxf(Mrun, mt);
      const float corr = __expf(Mrun - newM);
      acc0 *= corr; acc1 *= corr; acc2 *= corr; acc3 *= corr;
      Srun = Srun * corr; Mrun = newM;
      float ev[16];
      float ssum = 0.f;
#pragma unroll
      for (int k = 0; k < 16; ++k) {
        ev[k] = (lv[k] > -1e29f) ? __expf(lv[k] - Mrun) : 0.f;
        ssum += ev[k];
      }
      Srun += ssum;
#pragma unroll
      for (int v = 0; v < 8; ++v) wr[v] = hif ? ev[v + 8] : ev[v];
    }
    // GEMM4: g = relu(pe @ Wg_e + xg(i) + bg1) -> bG (16x64)
    {
      const v16h a = ldsA(bPe, 32, 0, lane);
#pragma unroll
      for (int t = 0; t < 4; ++t) {
        const v16h bb = ldsBt(Wget, 32, 0, 16 * t, lane);
        const int n = 16 * t + nloc;
        v8f c = wmma_f16(a, bb, splat8(bg1s[n] + xgs[n]));
#pragma unroll
        for (int v = 0; v < 8; ++v) {
          const int m = v + 8 * hif;
          const float val = c[v];
          bG[m * 64 + n] = (h16)(val > 0.f ? val : 0.f);
        }
      }
    }
    WAVE_SYNC();
    // GEMM5: gates = sigmoid(g @ Wg2 + bg2); accumulate msg
    {
      const v16h a0 = ldsA(bG, 64, 0, lane);
      const v16h a1 = ldsA(bG, 64, 32, lane);
      float accl[4] = {0.f, 0.f, 0.f, 0.f};
#pragma unroll
      for (int t = 0; t < 4; ++t) {
        const v16h b0 = ldsBt(Wg2t, 64, 0, 16 * t, lane);
        const v16h b1 = ldsBt(Wg2t, 64, 32, 16 * t, lane);
        const int o = 16 * t + nloc;
        v8f c = wmma_f16(a0, b0, splat8(bg2s[o]));
        c = wmma_f16(a1, b1, c);
#pragma unroll
        for (int v = 0; v < 8; ++v) {
          const int m = v + 8 * hif;
          const float gate = 1.f / (1.f + __expf(-c[v]));
          const float ntv = nt[((size_t)(b * Nn) + (j0 + m)) * 64 + o];
          accl[t] += wr[v] * gate * ntv;
        }
      }
      acc0 += accl[0]; acc1 += accl[1]; acc2 += accl[2]; acc3 += accl[3];
    }
  }

  // --- merge per-wave softmax states ---
  atomicAdd(&accW[wv * 64 + 0 * 16 + nloc], acc0);
  atomicAdd(&accW[wv * 64 + 1 * 16 + nloc], acc1);
  atomicAdd(&accW[wv * 64 + 2 * 16 + nloc], acc2);
  atomicAdd(&accW[wv * 64 + 3 * 16 + nloc], acc3);
  if (lane == 0) { Mw[wv] = Mrun; Sw[wv] = Srun; }
  __syncthreads();

  if (tid < 64) {
    const float Mg = fmaxf(fmaxf(Mw[0], Mw[1]), fmaxf(Mw[2], Mw[3]));
    float S = 0.f, m = 0.f;
#pragma unroll
    for (int w = 0; w < 4; ++w) {
      const float e = __expf(Mw[w] - Mg);
      S += Sw[w] * e;
      m += accW[w * 64 + tid] * e;
    }
    msgv[tid] = m / fmaxf(S, 1e-30f);
  }
  __syncthreads();

  // --- final combine: relu([sf, msg] @ Wc1 + bc1) @ Wc2 + bc2 ---
  if (tid < 64) {
    float s = bc1[tid];
#pragma unroll
    for (int k = 0; k < 64; ++k) s += sfs[k] * Wc1[k * 64 + tid];
#pragma unroll
    for (int k = 0; k < 64; ++k) s += msgv[k] * Wc1[(64 + k) * 64 + tid];
    t1v[tid] = s > 0.f ? s : 0.f;
  }
  __syncthreads();
  if (tid < 64) {
    float s = bc2[tid];
#pragma unroll
    for (int k = 0; k < 64; ++k) s += t1v[k] * Wc2[k * 64 + tid];
    out[row_i * 64 + tid] = s;
  }
}

extern "C" void kernel_launch(void* const* d_in, const int* in_sizes, int n_in,
                              void* d_out, int out_size, void* d_ws, size_t ws_size,
                              hipStream_t stream) {
  const float* x   = (const float*)d_in[0];
  const float* adj = (const float*)d_in[1];
  const float* ef  = (const float*)d_in[2];
  const float* Ws  = (const float*)d_in[3];
  const float* bs  = (const float*)d_in[4];
  const float* Wn  = (const float*)d_in[5];
  const float* bn  = (const float*)d_in[6];
  const float* We1 = (const float*)d_in[7];
  const float* be1 = (const float*)d_in[8];
  const float* We2 = (const float*)d_in[9];
  const float* be2 = (const float*)d_in[10];
  const float* Wa1 = (const float*)d_in[11];
  const float* ba1 = (const float*)d_in[12];
  const float* Wa2 = (const float*)d_in[13];
  const float* ba2 = (const float*)d_in[14];
  const float* Wg1 = (const float*)d_in[15];
  const float* bg1 = (const float*)d_in[16];
  const float* Wg2 = (const float*)d_in[17];
  const float* bg2 = (const float*)d_in[18];
  const float* Wc1 = (const float*)d_in[19];
  const float* bc1 = (const float*)d_in[20];
  const float* Wc2 = (const float*)d_in[21];
  const float* bc2 = (const float*)d_in[22];
  float* out = (float*)d_out;

  const int BN = Bn * Nn;
  float* ws  = (float*)d_ws;
  float* sfp = ws;                  // BN*64
  float* ntp = sfp + BN * 64;       // BN*64
  float* xai = ntp + BN * 64;       // BN*32
  float* xaj = xai + BN * 32;       // BN*32
  float* xgp = xaj + BN * 32;       // BN*64

  sg_pre<<<BN, 256, 0, stream>>>(x, Ws, bs, Wn, bn, Wa1, Wg1,
                                 sfp, ntp, xai, xaj, xgp);
  sg_main<<<dim3(Nn, Bn), 128, 0, stream>>>(
      adj, ef, We1, be1, We2, be2, Wa1, ba1, Wa2, ba2,
      Wg1, bg1, Wg2, bg2, Wc1, bc1, Wc2, bc2,
      sfp, ntp, xai, xaj, xgp, out);
}
